// KimiMoE_3032246911035
// MI455X (gfx1250) — compile-verified, matrix operand order checked by reading, not run on hardware
//
#include <hip/hip_runtime.h>
#include <hip/hip_bf16.h>
#include <cstdint>
#include <cstddef>

// ---------------------------------------------------------------------------
// Kimi-style MoE for MI455X (gfx1250, wave32, WMMA).
// Sparse top-2 routing with per-expert token lists; all GEMMs via
// v_wmma_f32_16x16x32_bf16 (fp32 accumulate). Per-wave tiles sized so that
// 2 accumulators (16 VGPR) + double-buffered fragments (2x3x8 VGPR) fit the
// ~64-VGPR allocation sweet spot.
// Inner loops are explicitly one-step software-pipelined.
// ---------------------------------------------------------------------------

typedef __bf16 bf16;
typedef __attribute__((ext_vector_type(16))) __bf16 v16bf;
typedef __attribute__((ext_vector_type(8)))  __bf16 v8bf;
typedef __attribute__((ext_vector_type(8)))  float  v8f;

static constexpr int T   = 2048;   // B*S tokens
static constexpr int D   = 1024;   // hidden
static constexpr int E   = 8;      // routed experts
static constexpr int I   = 1408;   // routed intermediate
static constexpr int ISH = 2816;   // shared intermediate (2*I)

// ----- WMMA helpers ---------------------------------------------------------
__device__ __forceinline__ v8f wmma_bf16(v16bf a, v16bf b, v8f c) {
  // D = A(16x32 bf16) * B(32x16 bf16) + C(16x16 f32)
  return __builtin_amdgcn_wmma_f32_16x16x32_bf16(
      /*neg_a=*/false, a, /*neg_b=*/false, b,
      /*c_mod=*/(short)0, c, /*reuse_a=*/false, /*reuse_b=*/false);
}

// Load one 16-bit fragment per the CDNA5 16-bit A striping (ISA 7.12.2):
// lane L holds row (tileRow + L%16); halves 0..7 = K (L/16)*8..+7,
// halves 8..15 = K 16+(L/16)*8..+7  => two contiguous 16-byte loads.
__device__ __forceinline__ v16bf load_frag(const bf16* __restrict__ rowbase,
                                           int koff) {
  union { v16bf v; v8bf h[2]; } u;
  const v8bf* p = reinterpret_cast<const v8bf*>(rowbase + koff);
  u.h[0] = p[0];
  u.h[1] = p[2];
  return u.v;
}

__device__ __forceinline__ float silu(float v) {
  return v / (1.0f + __expf(-v));
}

// ----- fp32 -> bf16 conversion ---------------------------------------------
__global__ void __launch_bounds__(256) cvt_f32_bf16(const float* __restrict__ in,
                                                    bf16* __restrict__ out,
                                                    size_t n) {
  size_t i = (size_t)blockIdx.x * blockDim.x + threadIdx.x;
  size_t stride = (size_t)gridDim.x * blockDim.x;
  for (; i < n; i += stride) out[i] = (bf16)in[i];
}

// ----- gate: sigmoid + bias, top-2, renorm*2.5, build expert lists ----------
__global__ void __launch_bounds__(256) gate_kernel(
    const float* __restrict__ x, const float* __restrict__ gw,
    const float* __restrict__ gbias,
    int* __restrict__ cnt, int* __restrict__ list, float* __restrict__ wl) {
  int t    = blockIdx.x;
  int tid  = threadIdx.x;
  int lane = tid & 31;
  int e    = tid >> 5;                       // 8 waves -> 8 experts
  const float* xr = x  + (size_t)t * D;
  const float* gr = gw + (size_t)e * D;
  float s = 0.f;
  for (int d = lane; d < D; d += 32) s += xr[d] * gr[d];
  for (int off = 16; off; off >>= 1) s += __shfl_xor(s, off, 32);
  __shared__ float slog[E];
  if (lane == 0) slog[e] = s;
  __syncthreads();
  if (tid == 0) {
    float choice[E];
    for (int i = 0; i < E; ++i)
      choice[i] = 1.f / (1.f + __expf(-slog[i])) + gbias[i];
    int i0 = 0;
    for (int i = 1; i < E; ++i) if (choice[i] > choice[i0]) i0 = i;  // first max
    int i1 = -1;
    for (int i = 0; i < E; ++i) {
      if (i == i0) continue;
      if (i1 < 0 || choice[i] > choice[i1]) i1 = i;
    }
    float w0 = choice[i0], w1 = choice[i1];
    float inv = 2.5f / (w0 + w1 + 1e-20f);
    w0 *= inv; w1 *= inv;
    int s0 = atomicAdd(&cnt[i0], 1); list[i0 * T + s0] = t; wl[i0 * T + s0] = w0;
    int s1 = atomicAdd(&cnt[i1], 1); list[i1 * T + s1] = t; wl[i1 * T + s1] = w1;
  }
}

// ----- routed up-proj + SwiGLU ---------------------------------------------
// grid = (T/32, I/64, E); block 256 = 8 waves (2M x 4N).
// Per wave: 16x16 v-tile + 16x16 g-tile sharing one A fragment
// (3 frags / 2 WMMAs per k-step), explicitly pipelined one step ahead.
__global__ void __launch_bounds__(256, 1) moe_up_routed(
    const bf16* __restrict__ xb, const bf16* __restrict__ w1b,
    const float* __restrict__ b1, const int* __restrict__ cnt,
    const int* __restrict__ list, bf16* __restrict__ hbuf) {
  int e  = blockIdx.z;
  int Ne = cnt[e];
  int M0 = blockIdx.x * 32;
  if (M0 >= Ne) return;
  __shared__ int s_tok[32];
  int tid = threadIdx.x;
  if (tid < 32) {
    int r = M0 + tid;
    s_tok[tid] = (r < Ne) ? list[e * T + r] : 0;
  }
  __syncthreads();
  int lane = tid & 31, wave = tid >> 5;
  int wm = wave >> 2, wn = wave & 3;
  int n0 = blockIdx.y * 64 + wn * 16;        // column within [0, I)
  int ksub = (lane >> 4) * 8;
  const bf16* aptr = xb + (size_t)s_tok[wm * 16 + (lane & 15)] * D;
  const bf16* bv   = w1b + ((size_t)e * 2 * I + (n0 + (lane & 15))) * D;
  const bf16* bg   = bv + (size_t)I * D;
  v8f accv = {}, accg = {};
  v16bf a_c  = load_frag(aptr, ksub);
  v16bf bv_c = load_frag(bv,   ksub);
  v16bf bg_c = load_frag(bg,   ksub);
  for (int k = 32; k < D; k += 32) {
    v16bf a_n  = load_frag(aptr, k + ksub);
    v16bf bv_n = load_frag(bv,   k + ksub);
    v16bf bg_n = load_frag(bg,   k + ksub);
    accv = wmma_bf16(a_c, bv_c, accv);
    accg = wmma_bf16(a_c, bg_c, accg);
    a_c = a_n; bv_c = bv_n; bg_c = bg_n;
  }
  accv = wmma_bf16(a_c, bv_c, accv);
  accg = wmma_bf16(a_c, bg_c, accg);

  int ncol = n0 + (lane & 15);
  const float* b1e = b1 + (size_t)e * 2 * I;
  float badd_v = b1e[ncol];
  float badd_g = b1e[I + ncol];
  int mo = M0 + wm * 16;
  for (int r = 0; r < 8; ++r) {
    int m = mo + r + 8 * (lane >> 4);        // slot row in expert list
    if (m < Ne) {
      float v = accv[r] + badd_v;
      float g = accg[r] + badd_g;
      hbuf[((size_t)e * T + m) * I + ncol] = (bf16)(silu(v) * g);
    }
  }
}

// ----- routed down-proj + weighted scatter ---------------------------------
// grid = (T/32, D/128, E); block 256 = 8 waves (2M x 4N), wave = 16M x 32N
// (1x2 tile: 3 frags / 2 WMMAs). routed[t][d] += w * (h@W2^T + b2)
__global__ void __launch_bounds__(256, 1) moe_down_routed(
    const bf16* __restrict__ hbuf, const bf16* __restrict__ w2b,
    const float* __restrict__ b2, const int* __restrict__ cnt,
    const int* __restrict__ list, const float* __restrict__ wl,
    float* __restrict__ routed) {
  int e  = blockIdx.z;
  int Ne = cnt[e];
  int M0 = blockIdx.x * 32;
  if (M0 >= Ne) return;
  __shared__ int   s_tok[32];
  __shared__ float s_w[32];
  int tid = threadIdx.x;
  if (tid < 32) {
    int r = M0 + tid;
    s_tok[tid] = (r < Ne) ? list[e * T + r] : 0;
    s_w[tid]   = (r < Ne) ? wl[e * T + r]   : 0.f;
  }
  __syncthreads();
  int lane = tid & 31, wave = tid >> 5;
  int wm = wave >> 2, wn = wave & 3;
  int mo = M0 + wm * 16;                     // wave's 16-row slot base
  int n0 = blockIdx.y * 128 + wn * 32;       // column pair base within [0, D)
  int ksub = (lane >> 4) * 8;
  const bf16* aptr = hbuf + ((size_t)e * T + (mo + (lane & 15))) * I;
  const bf16* b0   = w2b + ((size_t)e * D + (n0 + (lane & 15))) * I;
  const bf16* b1p  = b0 + (size_t)16 * I;
  v8f c0 = {}, c1 = {};
  v16bf a_c  = load_frag(aptr, ksub);
  v16bf b0_c = load_frag(b0,   ksub);
  v16bf b1_c = load_frag(b1p,  ksub);
  for (int k = 32; k < I; k += 32) {
    v16bf a_n  = load_frag(aptr, k + ksub);
    v16bf b0_n = load_frag(b0,   k + ksub);
    v16bf b1_n = load_frag(b1p,  k + ksub);
    c0 = wmma_bf16(a_c, b0_c, c0);
    c1 = wmma_bf16(a_c, b1_c, c1);
    a_c = a_n; b0_c = b0_n; b1_c = b1_n;
  }
  c0 = wmma_bf16(a_c, b0_c, c0);
  c1 = wmma_bf16(a_c, b1_c, c1);

  int nc0 = n0 + (lane & 15);
  int nc1 = nc0 + 16;
  float bias0 = b2[(size_t)e * D + nc0];
  float bias1 = b2[(size_t)e * D + nc1];
  for (int r = 0; r < 8; ++r) {
    int m = mo + r + 8 * (lane >> 4);
    if (m < Ne) {
      int li = m - M0;
      float w = s_w[li];
      float* rr = routed + (size_t)s_tok[li] * D;
      atomicAdd(&rr[nc0], (c0[r] + bias0) * w);
      atomicAdd(&rr[nc1], (c1[r] + bias1) * w);
    }
  }
}

// ----- shared expert up-proj + SwiGLU (dense) -------------------------------
// grid = (T/32, ISH/64, 1); same tiling as moe_up_routed, no gather.
__global__ void __launch_bounds__(256, 1) moe_up_shared(
    const bf16* __restrict__ xb, const bf16* __restrict__ ws1b,
    const float* __restrict__ bs1, bf16* __restrict__ hsh) {
  int tid = threadIdx.x;
  int lane = tid & 31, wave = tid >> 5;
  int wm = wave >> 2, wn = wave & 3;
  int mo = blockIdx.x * 32 + wm * 16;        // token row (64*32 = T, all valid)
  int n0 = blockIdx.y * 64 + wn * 16;        // column within [0, ISH)
  int ksub = (lane >> 4) * 8;
  const bf16* aptr = xb + (size_t)(mo + (lane & 15)) * D;
  const bf16* bv   = ws1b + (size_t)(n0 + (lane & 15)) * D;
  const bf16* bg   = bv + (size_t)ISH * D;
  v8f accv = {}, accg = {};
  v16bf a_c  = load_frag(aptr, ksub);
  v16bf bv_c = load_frag(bv,   ksub);
  v16bf bg_c = load_frag(bg,   ksub);
  for (int k = 32; k < D; k += 32) {
    v16bf a_n  = load_frag(aptr, k + ksub);
    v16bf bv_n = load_frag(bv,   k + ksub);
    v16bf bg_n = load_frag(bg,   k + ksub);
    accv = wmma_bf16(a_c, bv_c, accv);
    accg = wmma_bf16(a_c, bg_c, accg);
    a_c = a_n; bv_c = bv_n; bg_c = bg_n;
  }
  accv = wmma_bf16(a_c, bv_c, accv);
  accg = wmma_bf16(a_c, bg_c, accg);

  int ncol = n0 + (lane & 15);
  float badd_v = bs1[ncol];
  float badd_g = bs1[ISH + ncol];
  for (int r = 0; r < 8; ++r) {
    int m = mo + r + 8 * (lane >> 4);
    float v = accv[r] + badd_v;
    float g = accg[r] + badd_g;
    hsh[(size_t)m * ISH + ncol] = (bf16)(silu(v) * g);
  }
}

// ----- shared expert down-proj + final combine ------------------------------
// grid = (T/32, D/128, 1); wave = 16M x 32N. out = hsh@Ws2^T + bs2 + routed
__global__ void __launch_bounds__(256, 1) moe_down_shared_final(
    const bf16* __restrict__ hsh, const bf16* __restrict__ ws2b,
    const float* __restrict__ bs2, const float* __restrict__ routed,
    float* __restrict__ out) {
  int tid = threadIdx.x;
  int lane = tid & 31, wave = tid >> 5;
  int wm = wave >> 2, wn = wave & 3;
  int mo = blockIdx.x * 32 + wm * 16;
  int n0 = blockIdx.y * 128 + wn * 32;
  int ksub = (lane >> 4) * 8;
  const bf16* aptr = hsh  + (size_t)(mo + (lane & 15)) * ISH;
  const bf16* b0   = ws2b + (size_t)(n0 + (lane & 15)) * ISH;
  const bf16* b1p  = b0 + (size_t)16 * ISH;
  v8f c0 = {}, c1 = {};
  v16bf a_c  = load_frag(aptr, ksub);
  v16bf b0_c = load_frag(b0,   ksub);
  v16bf b1_c = load_frag(b1p,  ksub);
  for (int k = 32; k < ISH; k += 32) {
    v16bf a_n  = load_frag(aptr, k + ksub);
    v16bf b0_n = load_frag(b0,   k + ksub);
    v16bf b1_n = load_frag(b1p,  k + ksub);
    c0 = wmma_bf16(a_c, b0_c, c0);
    c1 = wmma_bf16(a_c, b1_c, c1);
    a_c = a_n; b0_c = b0_n; b1_c = b1_n;
  }
  c0 = wmma_bf16(a_c, b0_c, c0);
  c1 = wmma_bf16(a_c, b1_c, c1);

  int nc0 = n0 + (lane & 15);
  int nc1 = nc0 + 16;
  float bias0 = bs2[nc0];
  float bias1 = bs2[nc1];
  for (int r = 0; r < 8; ++r) {
    int m = mo + r + 8 * (lane >> 4);
    size_t i0 = (size_t)m * D + nc0;
    size_t i1 = (size_t)m * D + nc1;
    out[i0] = c0[r] + bias0 + routed[i0];
    out[i1] = c1[r] + bias1 + routed[i1];
  }
}

// ---------------------------------------------------------------------------
static inline size_t align_up(size_t v, size_t a) { return (v + a - 1) & ~(a - 1); }

extern "C" void kernel_launch(void* const* d_in, const int* in_sizes, int n_in,
                              void* d_out, int out_size, void* d_ws, size_t ws_size,
                              hipStream_t stream) {
  (void)in_sizes; (void)n_in; (void)out_size; (void)ws_size;
  const float* x     = (const float*)d_in[0];
  const float* gw    = (const float*)d_in[1];
  const float* gbias = (const float*)d_in[2];
  const float* W1    = (const float*)d_in[3];
  const float* b1    = (const float*)d_in[4];
  const float* W2    = (const float*)d_in[5];
  const float* b2    = (const float*)d_in[6];
  const float* Ws1   = (const float*)d_in[7];
  const float* bs1   = (const float*)d_in[8];
  const float* Ws2   = (const float*)d_in[9];
  const float* bs2   = (const float*)d_in[10];
  float* out = (float*)d_out;

  // ---- workspace carve (~157 MB total) ----
  const size_t n_x   = (size_t)T * D;
  const size_t n_w1  = (size_t)E * 2 * I * D;
  const size_t n_w2  = (size_t)E * D * I;
  const size_t n_ws1 = (size_t)2 * ISH * D;
  const size_t n_ws2 = (size_t)D * ISH;

  uint8_t* p = (uint8_t*)d_ws;
  size_t off = 0;
  auto carve = [&](size_t bytes) {
    uint8_t* q = p + off;
    off = align_up(off + bytes, 256);
    return q;
  };
  bf16*  xb     = (bf16*)carve(n_x   * sizeof(bf16));
  bf16*  w1b    = (bf16*)carve(n_w1  * sizeof(bf16));
  bf16*  w2b    = (bf16*)carve(n_w2  * sizeof(bf16));
  bf16*  ws1b   = (bf16*)carve(n_ws1 * sizeof(bf16));
  bf16*  ws2b   = (bf16*)carve(n_ws2 * sizeof(bf16));
  bf16*  hbuf   = (bf16*)carve((size_t)E * T * I * sizeof(bf16));
  bf16*  hsh    = (bf16*)carve((size_t)T * ISH * sizeof(bf16));
  float* routed = (float*)carve((size_t)T * D * sizeof(float));
  int*   cnt    = (int*)carve(E * sizeof(int));
  int*   list   = (int*)carve((size_t)E * T * sizeof(int));
  float* wl     = (float*)carve((size_t)E * T * sizeof(float));

  hipMemsetAsync(cnt, 0, E * sizeof(int), stream);
  hipMemsetAsync(routed, 0, (size_t)T * D * sizeof(float), stream);

  // ---- fp32 -> bf16 conversions ----
  cvt_f32_bf16<<<4096, 256, 0, stream>>>(x,   xb,   n_x);
  cvt_f32_bf16<<<4096, 256, 0, stream>>>(W1,  w1b,  n_w1);
  cvt_f32_bf16<<<4096, 256, 0, stream>>>(W2,  w2b,  n_w2);
  cvt_f32_bf16<<<4096, 256, 0, stream>>>(Ws1, ws1b, n_ws1);
  cvt_f32_bf16<<<4096, 256, 0, stream>>>(Ws2, ws2b, n_ws2);

  // ---- gate + routing lists ----
  gate_kernel<<<T, 256, 0, stream>>>(x, gw, gbias, cnt, list, wl);

  // ---- routed experts (sparse) ----
  moe_up_routed<<<dim3(T / 32, I / 64, E), 256, 0, stream>>>(
      xb, w1b, b1, cnt, list, hbuf);
  moe_down_routed<<<dim3(T / 32, D / 128, E), 256, 0, stream>>>(
      hbuf, w2b, b2, cnt, list, wl, routed);

  // ---- shared expert + final combine ----
  moe_up_shared<<<dim3(T / 32, ISH / 64, 1), 256, 0, stream>>>(
      xb, ws1b, bs1, hsh);
  moe_down_shared_final<<<dim3(T / 32, D / 128, 1), 256, 0, stream>>>(
      hsh, ws2b, bs2, routed, out);
}
